// MuZeroNet_80642305950625
// MI455X (gfx1250) — compile-verified, matrix operand order checked by reading, not run on hardware
//
#include <hip/hip_runtime.h>
#include <cstdint>
#include <cstddef>

// ---------------------------------------------------------------------------
// MuZero-style network forward for MI455X (gfx1250, wave32, WMMA).
//
// Compute:  ~1.16 TFLOP of GEMMs at B=8192  -> bf16 WMMA (v_wmma_f32_16x16x32_bf16)
// Memory:   ~140 MB weights + activations   -> 23.3 TB/s HBM, L2 192MB
// GEMM tile: block 128x128xK, BK=32, 256 thr = 8 waves, wave tile 64x32
//            (4 x 2 WMMA 16x16 frags, fp32 accum = 8 v8f = 64 VGPRs/wave)
// Staging:  double-buffered GLOBAL_LOAD_ASYNC_TO_LDS_B128 (ASYNCcnt) when the
//           toolchain exposes the gfx1250 async builtins, else VGPR+ds_store.
//
// Input flattening assumption (JAX pytree: dicts sorted by key, recursively):
//  0 obs
//  1 pol_l1.b   2 pol_l1.w   3 pol_l2.b   4 pol_l2.w   5 pol_l3.b   6 pol_l3.w
//  7 pol_ln1.b  8 pol_ln1.g  9 pol_ln2.b 10 pol_ln2.g
// 11 policy_moe.b1 12 .b2 13 .gate.b 14 .gate.w 15 .w1 16 .w2
// 17 rep_l1.b 18 rep_l1.w 19 rep_l2.b 20 rep_l2.w 21 rep_l3.b 22 rep_l3.w
// 23 rep_ln1.b 24 rep_ln1.g 25 rep_ln2.b 26 rep_ln2.g 27 rep_ln3.b 28 rep_ln3.g
// 29 rep_moe.b1 30 .b2 31 .gate.b 32 .gate.w 33 .w1 34 .w2
// 35..42 rep_rb1{fc1.b,fc1.w,fc2.b,fc2.w,ln1.b,ln1.g,ln2.b,ln2.g}
// 43..50 rep_rb2{...} 51..58 rep_rb3{...} 59..66 rep_rb4{...}
// 67 val_l1.b 68 val_l1.w 69 val_l2.b 70 val_l2.w 71 val_l3.b 72 val_l3.w
// 73 val_ln1.b 74 val_ln1.g 75 val_ln2.b 76 val_ln2.g
//
// d_out: [policy 8192*64 f32][value 8192 f32][h 8192*1024 f32]
// ---------------------------------------------------------------------------

typedef __attribute__((ext_vector_type(16))) __bf16 v16bf;
typedef __attribute__((ext_vector_type(8)))  __bf16 v8bf;
typedef __attribute__((ext_vector_type(8)))  float  v8f;

#if defined(__has_builtin)
#  if __has_builtin(__builtin_amdgcn_global_load_async_to_lds_b128) && \
      __has_builtin(__builtin_amdgcn_s_wait_asynccnt)
#    define USE_ASYNC_LDS 1
#  endif
#endif
#ifndef USE_ASYNC_LDS
#  define USE_ASYNC_LDS 0
#endif

#if USE_ASYNC_LDS
// The builtin takes pointers to int __attribute__((vector_size(16))) in
// addrspace(1) (src) and addrspace(3) (dst) -- per hipcc diagnostics.
typedef int v4i_vs __attribute__((vector_size(16)));
#define AS1 __attribute__((address_space(1)))
#define AS3 __attribute__((address_space(3)))
// Flat->global: numerically identical on AMDGPU.
__device__ __forceinline__ AS1 v4i_vs* to_global(const void* p)
{
    return (AS1 v4i_vs*)(unsigned long long)(uintptr_t)p;
}
// Flat shared address: low 32 bits are the LDS byte offset (ISA 10.2 aperture).
__device__ __forceinline__ AS3 v4i_vs* to_lds(const void* p)
{
    return (AS3 v4i_vs*)(unsigned)(uintptr_t)p;
}
#endif

// ---------------------------------------------------------------------------
// Utility kernels
// ---------------------------------------------------------------------------

__global__ __launch_bounds__(256)
void cast_f32_bf16_kernel(const float* __restrict__ in, __bf16* __restrict__ out, size_t n)
{
    size_t i = (size_t)blockIdx.x * 256 + threadIdx.x;
    if (i < n) out[i] = (__bf16)in[i];
}

// W: [K][N] f32 row-major  ->  Wt: [N][K] bf16 row-major
__global__ __launch_bounds__(256)
void transpose_cast_kernel(const float* __restrict__ W, __bf16* __restrict__ Wt, int K, int N)
{
    __shared__ float tile[32][33];
    const int nb = blockIdx.x * 32;
    const int kb = blockIdx.y * 32;
    const int tx = threadIdx.x & 31;
    const int ty = threadIdx.x >> 5;        // 0..7
    #pragma unroll
    for (int i = ty; i < 32; i += 8) {
        int k = kb + i, n = nb + tx;
        tile[i][tx] = (k < K && n < N) ? W[(size_t)k * N + n] : 0.0f;
    }
    __syncthreads();
    #pragma unroll
    for (int i = ty; i < 32; i += 8) {
        int n = nb + i, k = kb + tx;
        if (n < N && k < K) Wt[(size_t)n * K + k] = (__bf16)tile[tx][i];
    }
}

// ---------------------------------------------------------------------------
// bf16 WMMA GEMM:  out[M,N] = epilogue(A[M,K] @ Wt[N,K]^T + bias)
// ACT: 0=none 1=exact GELU; RES: add resid; WBF/WF32: bf16/f32 output writes
// Requirements: M % 128 == 0, K % 32 == 0, Wt allocated with rows padded to a
// multiple of 128 (padding rows may be garbage; their columns are never stored).
// ---------------------------------------------------------------------------
template<int ACT, int RES, int WBF, int WF32>
__global__ __launch_bounds__(256)
void gemm_bf16_wmma(const __bf16* __restrict__ A,
                    const __bf16* __restrict__ Bt,
                    const float*  __restrict__ bias,
                    const __bf16* __restrict__ resid,
                    __bf16* __restrict__ outb,
                    float*  __restrict__ outf,
                    int M, int N, int K)
{
#if USE_ASYNC_LDS
    __shared__ alignas(16) __bf16 As[2][128][40];
    __shared__ alignas(16) __bf16 Bs[2][128][40];
#else
    __shared__ alignas(16) __bf16 As[1][128][40];
    __shared__ alignas(16) __bf16 Bs[1][128][40];
#endif

    const int tid  = threadIdx.x;
    const int wave = tid >> 5;            // 0..7
    const int lane = tid & 31;
    const int wm   = wave >> 2;           // 0..1 : wave row   (64 rows each)
    const int wn   = wave & 3;            // 0..3 : wave col   (32 cols each)
    const int half = lane >> 4;           // K-chunk selector per ISA layout
    const int r    = lane & 15;           // row/col within 16x16 frag

    const int mBase = blockIdx.y * 128;
    const int nBase = blockIdx.x * 128;

    v8f acc[4][2];
    #pragma unroll
    for (int im = 0; im < 4; ++im)
        #pragma unroll
        for (int in = 0; in < 2; ++in) {
            v8f z = {0.f,0.f,0.f,0.f,0.f,0.f,0.f,0.f};
            acc[im][in] = z;
        }

    // Fragment gather + 8 WMMAs on one LDS buffer.
    // ISA 16-bit A layout: lane holds K=[8*half,+8) in regs 0..3 and
    // K=[16+8*half,+8) in regs 4..7; B (pre-transposed) is symmetric.
    auto do_tile = [&](int b) {
        v16bf afr[4];
        #pragma unroll
        for (int im = 0; im < 4; ++im) {
            int row = wm * 64 + im * 16 + r;
            v8bf lo = *(const v8bf*)&As[b][row][half * 8];
            v8bf hi = *(const v8bf*)&As[b][row][16 + half * 8];
            #pragma unroll
            for (int j = 0; j < 8; ++j) { afr[im][j] = lo[j]; afr[im][8 + j] = hi[j]; }
        }
        v16bf bfr[2];
        #pragma unroll
        for (int in = 0; in < 2; ++in) {
            int col = wn * 32 + in * 16 + r;
            v8bf lo = *(const v8bf*)&Bs[b][col][half * 8];
            v8bf hi = *(const v8bf*)&Bs[b][col][16 + half * 8];
            #pragma unroll
            for (int j = 0; j < 8; ++j) { bfr[in][j] = lo[j]; bfr[in][8 + j] = hi[j]; }
        }
        #pragma unroll
        for (int im = 0; im < 4; ++im)
            #pragma unroll
            for (int in = 0; in < 2; ++in)
                acc[im][in] = __builtin_amdgcn_wmma_f32_16x16x32_bf16(
                    false, afr[im], false, bfr[in],
                    (short)0, acc[im][in], false, false);
    };

#if USE_ASYNC_LDS
    // Double-buffered async global->LDS pipeline (ASYNCcnt).
    const __bf16* Ag = A  + (size_t)mBase * K;
    const __bf16* Bg = Bt + (size_t)nBase * K;
    auto stage = [&](int b, int k0) {
        #pragma unroll
        for (int i = 0; i < 2; ++i) {
            int cid = tid + i * 256;          // 0..511
            int rr  = cid >> 2;               // 0..127
            int cc  = (cid & 3) << 3;         // 0,8,16,24
            __builtin_amdgcn_global_load_async_to_lds_b128(
                to_global(&Ag[(size_t)rr * K + (k0 + cc)]),
                to_lds(&As[b][rr][cc]), 0, 0);
            __builtin_amdgcn_global_load_async_to_lds_b128(
                to_global(&Bg[(size_t)rr * K + (k0 + cc)]),
                to_lds(&Bs[b][rr][cc]), 0, 0);
        }
    };
    stage(0, 0);
    int buf = 0;
    for (int k0 = 0; k0 < K; k0 += 32) {
        __builtin_amdgcn_s_wait_asynccnt(0);   // own stage for `buf` landed
        __syncthreads();                        // everyone's landed; buf^1 free
        if (k0 + 32 < K) stage(buf ^ 1, k0 + 32);
        do_tile(buf);
        buf ^= 1;
    }
#else
    // Fallback: single-buffer VGPR staging.
    for (int k0 = 0; k0 < K; k0 += 32) {
        #pragma unroll
        for (int i = 0; i < 2; ++i) {
            int cid = tid + i * 256;
            int rr  = cid >> 2;
            int cc  = (cid & 3) << 3;
            *(uint4*)&As[0][rr][cc] =
                *(const uint4*)&A[(size_t)(mBase + rr) * K + (k0 + cc)];
            *(uint4*)&Bs[0][rr][cc] =
                *(const uint4*)&Bt[(size_t)(nBase + rr) * K + (k0 + cc)];
        }
        __syncthreads();
        do_tile(0);
        __syncthreads();
    }
#endif

    // Epilogue. C layout: lanes 0-15 -> M=v,   N=lane
    //                     lanes16-31 -> M=v+8, N=lane-16
    #pragma unroll
    for (int in = 0; in < 2; ++in) {
        int n = nBase + wn * 32 + in * 16 + r;
        if (n < N) {
            float bsv = bias[n];
            #pragma unroll
            for (int im = 0; im < 4; ++im) {
                int mrow = mBase + wm * 64 + im * 16 + half * 8;
                #pragma unroll
                for (int v = 0; v < 8; ++v) {
                    int m = mrow + v;
                    float x = acc[im][in][v] + bsv;
                    if (ACT == 1)
                        x = 0.5f * x * (1.0f + erff(x * 0.70710678118654752f));
                    if (RES)  x += (float)resid[(size_t)m * N + n];
                    if (WBF)  outb[(size_t)m * N + n] = (__bf16)x;
                    if (WF32) outf[(size_t)m * N + n] = x;
                }
            }
        }
    }
}

// ---------------------------------------------------------------------------
// LayerNorm (one block per row), fp32 math, bf16 in/out, eps = 1e-5
// ---------------------------------------------------------------------------
__global__ __launch_bounds__(256)
void layernorm_kernel(const __bf16* __restrict__ x, const float* __restrict__ beta,
                      const float* __restrict__ gamma, __bf16* __restrict__ y, int D)
{
    const int row = blockIdx.x;
    const __bf16* xr = x + (size_t)row * D;
    float s = 0.f, ss = 0.f;
    for (int i = threadIdx.x; i < D; i += 256) {
        float v = (float)xr[i];
        s += v; ss += v * v;
    }
    #pragma unroll
    for (int off = 16; off > 0; off >>= 1) {
        s  += __shfl_xor(s,  off, 32);
        ss += __shfl_xor(ss, off, 32);
    }
    __shared__ float shs[8], shss[8];
    const int wave = threadIdx.x >> 5, lane = threadIdx.x & 31;
    if (lane == 0) { shs[wave] = s; shss[wave] = ss; }
    __syncthreads();
    float ts = 0.f, tss = 0.f;
    #pragma unroll
    for (int i = 0; i < 8; ++i) { ts += shs[i]; tss += shss[i]; }
    const float mean = ts / (float)D;
    const float var  = tss / (float)D - mean * mean;
    const float inv  = rsqrtf(var + 1e-5f);
    __bf16* yr = y + (size_t)row * D;
    for (int i = threadIdx.x; i < D; i += 256)
        yr[i] = (__bf16)(((float)xr[i] - mean) * inv * gamma[i] + beta[i]);
}

// ---------------------------------------------------------------------------
// MoE gate logits: one wave per row, 4 experts, fp32 weights
// ---------------------------------------------------------------------------
__global__ __launch_bounds__(256)
void gate_kernel(const __bf16* __restrict__ x, const float* __restrict__ gw,
                 const float* __restrict__ gb, float* __restrict__ logits, int D)
{
    const int wave = threadIdx.x >> 5, lane = threadIdx.x & 31;
    const int row  = blockIdx.x * 8 + wave;
    const __bf16* xr = x + (size_t)row * D;
    float a[4] = {0.f, 0.f, 0.f, 0.f};
    for (int k = lane; k < D; k += 32) {
        float v = (float)xr[k];
        #pragma unroll
        for (int e = 0; e < 4; ++e) a[e] += v * gw[k * 4 + e];
    }
    #pragma unroll
    for (int e = 0; e < 4; ++e)
        #pragma unroll
        for (int off = 16; off > 0; off >>= 1) a[e] += __shfl_xor(a[e], off, 32);
    if (lane == 0) {
        #pragma unroll
        for (int e = 0; e < 4; ++e) logits[(size_t)row * 4 + e] = a[e] + gb[e];
    }
}

// ---------------------------------------------------------------------------
// MoE combine: softmax over 4 logits, top-2 renorm, weighted sum + residual
// ey layout: [e][B][D]
// ---------------------------------------------------------------------------
__global__ __launch_bounds__(256)
void moe_combine_kernel(const float* __restrict__ logits, const __bf16* __restrict__ ey,
                        const __bf16* __restrict__ x, __bf16* __restrict__ out,
                        int B, int D)
{
    size_t idx = (size_t)blockIdx.x * 256 + threadIdx.x;
    if (idx >= (size_t)B * D) return;
    const int b = (int)(idx / D);
    const int d = (int)(idx % D);
    float lg[4];
    #pragma unroll
    for (int e = 0; e < 4; ++e) lg[e] = logits[(size_t)b * 4 + e];
    float mx = fmaxf(fmaxf(lg[0], lg[1]), fmaxf(lg[2], lg[3]));
    float pe[4]; float sum = 0.f;
    #pragma unroll
    for (int e = 0; e < 4; ++e) { pe[e] = expf(lg[e] - mx); sum += pe[e]; }
    #pragma unroll
    for (int e = 0; e < 4; ++e) pe[e] /= sum;
    int i0 = 0;
    #pragma unroll
    for (int e = 1; e < 4; ++e) if (pe[e] > pe[i0]) i0 = e;
    int i1 = (i0 == 0) ? 1 : 0;
    #pragma unroll
    for (int e = 0; e < 4; ++e) if (e != i0 && pe[e] > pe[i1]) i1 = e;
    const float q  = pe[i0] + pe[i1];
    const float p0 = pe[i0] / q, p1 = pe[i1] / q;
    const float y0 = (float)ey[((size_t)i0 * B + b) * D + d];
    const float y1 = (float)ey[((size_t)i1 * B + b) * D + d];
    out[idx] = (__bf16)(p0 * y0 + p1 * y1 + (float)x[idx]);
}

// ---------------------------------------------------------------------------
// Value head: dot(row, w[256]) + b, tanh, fp32 out. One wave per row.
// ---------------------------------------------------------------------------
__global__ __launch_bounds__(256)
void value_head_kernel(const __bf16* __restrict__ v2, const float* __restrict__ w,
                       const float* __restrict__ bias, float* __restrict__ out, int D)
{
    const int wave = threadIdx.x >> 5, lane = threadIdx.x & 31;
    const int row  = blockIdx.x * 8 + wave;
    const __bf16* xr = v2 + (size_t)row * D;
    float a = 0.f;
    for (int k = lane; k < D; k += 32) a += (float)xr[k] * w[k];
    #pragma unroll
    for (int off = 16; off > 0; off >>= 1) a += __shfl_xor(a, off, 32);
    if (lane == 0) out[row] = tanhf(a + bias[0]);
}

// ---------------------------------------------------------------------------
// Host-side launch helpers
// ---------------------------------------------------------------------------
static inline void launch_gemm(hipStream_t s, int act, bool res, bool wbf, bool wf32,
                               const void* A, const void* Bt, const float* bias,
                               const void* R, void* ob, float* of, int M, int N, int K)
{
    dim3 grid((unsigned)((N + 127) / 128), (unsigned)(M / 128));
    dim3 block(256);
    const __bf16* a = (const __bf16*)A;
    const __bf16* b = (const __bf16*)Bt;
    const __bf16* r = (const __bf16*)R;
    __bf16* o = (__bf16*)ob;
    if (act == 1)            gemm_bf16_wmma<1,0,1,0><<<grid, block, 0, s>>>(a, b, bias, r, o, of, M, N, K);
    else if (res && wf32)    gemm_bf16_wmma<0,1,1,1><<<grid, block, 0, s>>>(a, b, bias, r, o, of, M, N, K);
    else if (res)            gemm_bf16_wmma<0,1,1,0><<<grid, block, 0, s>>>(a, b, bias, r, o, of, M, N, K);
    else if (wf32 && !wbf)   gemm_bf16_wmma<0,0,0,1><<<grid, block, 0, s>>>(a, b, bias, r, o, of, M, N, K);
    else                     gemm_bf16_wmma<0,0,1,0><<<grid, block, 0, s>>>(a, b, bias, r, o, of, M, N, K);
}

static inline void launch_tr(hipStream_t s, const float* W, void* Wt, int K, int N)
{
    dim3 grid((unsigned)((N + 31) / 32), (unsigned)((K + 31) / 32));
    transpose_cast_kernel<<<grid, dim3(256), 0, s>>>(W, (__bf16*)Wt, K, N);
}

static inline void launch_ln(hipStream_t s, const void* x, const float* beta,
                             const float* gamma, void* y, int rows, int D)
{
    layernorm_kernel<<<dim3((unsigned)rows), dim3(256), 0, s>>>(
        (const __bf16*)x, beta, gamma, (__bf16*)y, D);
}

// ---------------------------------------------------------------------------
extern "C" void kernel_launch(void* const* d_in, const int* in_sizes, int n_in,
                              void* d_out, int out_size, void* d_ws, size_t ws_size,
                              hipStream_t stream)
{
    (void)in_sizes; (void)n_in; (void)out_size; (void)ws_size;
    const int B = 8192;
    auto in = [&](int i) { return (const float*)d_in[i]; };

    // Bump allocator over d_ws (256B aligned sub-buffers).
    size_t cur = 0;
    auto alloc = [&](size_t bytes) -> void* {
        cur = (cur + 255) & ~(size_t)255;
        void* p = (char*)d_ws + cur;
        cur += bytes;
        return p;
    };

    // Transposed bf16 weights  Wt[N][K]  (row counts padded to 128 multiples)
    void* w_rep_l1t  = alloc((size_t)2048 * 4608 * 2);
    void* w_rb1_fc1t = alloc((size_t)4096 * 2048 * 2);
    void* w_rb1_fc2t = alloc((size_t)2048 * 4096 * 2);
    void* w_rep_l2t  = alloc((size_t)1536 * 2048 * 2);
    void* w_rb2_fc1t = alloc((size_t)3072 * 1536 * 2);
    void* w_rb2_fc2t = alloc((size_t)1536 * 3072 * 2);
    void* w_rep_l3t  = alloc((size_t)1024 * 1536 * 2);
    void* w_rb3_fc1t = alloc((size_t)2048 * 1024 * 2);
    void* w_rb3_fc2t = alloc((size_t)1024 * 2048 * 2);
    void* w_rb4_fc1t = alloc((size_t)2048 * 1024 * 2);
    void* w_rb4_fc2t = alloc((size_t)1024 * 2048 * 2);
    void* w_rmoe_w1t = alloc((size_t)4 * 1536 * 1024 * 2);
    void* w_rmoe_w2t = alloc((size_t)4 * 1024 * 1536 * 2);
    void* w_pmoe_w1t = alloc((size_t)4 * 1024 * 1024 * 2);
    void* w_pmoe_w2t = alloc((size_t)4 * 1024 * 1024 * 2);
    void* w_pol_l1t  = alloc((size_t)512 * 1024 * 2);
    void* w_pol_l2t  = alloc((size_t)256 * 512 * 2);
    void* w_pol_l3t  = alloc((size_t)128 * 256 * 2);   // N=64 padded to 128 rows
    void* w_val_l1t  = alloc((size_t)512 * 1024 * 2);
    void* w_val_l2t  = alloc((size_t)256 * 512 * 2);

    // Activations (bf16) + gate logits (fp32)
    void*  OBS = alloc((size_t)B * 4608 * 2);
    void*  X   = alloc((size_t)B * 4096 * 2);   // main / residual buffer
    void*  T1  = alloc((size_t)B * 4096 * 2);
    void*  T2  = alloc((size_t)B * 4096 * 2);
    void*  H   = alloc((size_t)B * 1024 * 2);   // latent h (kept for both heads)
    void*  EH  = alloc((size_t)4 * B * 1536 * 2);
    void*  EY  = alloc((size_t)4 * B * 1024 * 2);
    float* GL  = (float*)alloc((size_t)B * 4 * 4);

    // ---- weight prep (deterministic each call) ----
    launch_tr(stream, in(18), w_rep_l1t, 4608, 2048);
    launch_tr(stream, in(36), w_rb1_fc1t, 2048, 4096);
    launch_tr(stream, in(38), w_rb1_fc2t, 4096, 2048);
    launch_tr(stream, in(20), w_rep_l2t, 2048, 1536);
    launch_tr(stream, in(44), w_rb2_fc1t, 1536, 3072);
    launch_tr(stream, in(46), w_rb2_fc2t, 3072, 1536);
    launch_tr(stream, in(22), w_rep_l3t, 1536, 1024);
    launch_tr(stream, in(52), w_rb3_fc1t, 1024, 2048);
    launch_tr(stream, in(54), w_rb3_fc2t, 2048, 1024);
    launch_tr(stream, in(60), w_rb4_fc1t, 1024, 2048);
    launch_tr(stream, in(62), w_rb4_fc2t, 2048, 1024);
    for (int e = 0; e < 4; ++e) {
        launch_tr(stream, in(33) + (size_t)e * 1024 * 1536,
                  (char*)w_rmoe_w1t + (size_t)e * 1536 * 1024 * 2, 1024, 1536);
        launch_tr(stream, in(34) + (size_t)e * 1536 * 1024,
                  (char*)w_rmoe_w2t + (size_t)e * 1024 * 1536 * 2, 1536, 1024);
        launch_tr(stream, in(15) + (size_t)e * 1024 * 1024,
                  (char*)w_pmoe_w1t + (size_t)e * 1024 * 1024 * 2, 1024, 1024);
        launch_tr(stream, in(16) + (size_t)e * 1024 * 1024,
                  (char*)w_pmoe_w2t + (size_t)e * 1024 * 1024 * 2, 1024, 1024);
    }
    launch_tr(stream, in(2),  w_pol_l1t, 1024, 512);
    launch_tr(stream, in(4),  w_pol_l2t, 512, 256);
    launch_tr(stream, in(6),  w_pol_l3t, 256, 64);
    launch_tr(stream, in(68), w_val_l1t, 1024, 512);
    launch_tr(stream, in(70), w_val_l2t, 512, 256);

    {   // obs -> bf16
        size_t n = (size_t)B * 4608;
        cast_f32_bf16_kernel<<<dim3((unsigned)((n + 255) / 256)), dim3(256), 0, stream>>>(
            in(0), (__bf16*)OBS, n);
    }

    // ---- representation trunk ----
    launch_gemm(stream, 1, false, true, false, OBS, w_rep_l1t, in(17), nullptr, T1, nullptr, B, 2048, 4608);
    launch_ln(stream, T1, in(23), in(24), X, B, 2048);

    launch_ln(stream, X, in(39), in(40), T1, B, 2048);                                   // rb1
    launch_gemm(stream, 1, false, true, false, T1, w_rb1_fc1t, in(35), nullptr, T2, nullptr, B, 4096, 2048);
    launch_ln(stream, T2, in(41), in(42), T1, B, 4096);
    launch_gemm(stream, 0, true, true, false, T1, w_rb1_fc2t, in(37), X, X, nullptr, B, 2048, 4096);

    launch_gemm(stream, 1, false, true, false, X, w_rep_l2t, in(19), nullptr, T1, nullptr, B, 1536, 2048);
    launch_ln(stream, T1, in(25), in(26), X, B, 1536);

    launch_ln(stream, X, in(47), in(48), T1, B, 1536);                                   // rb2
    launch_gemm(stream, 1, false, true, false, T1, w_rb2_fc1t, in(43), nullptr, T2, nullptr, B, 3072, 1536);
    launch_ln(stream, T2, in(49), in(50), T1, B, 3072);
    launch_gemm(stream, 0, true, true, false, T1, w_rb2_fc2t, in(45), X, X, nullptr, B, 1536, 3072);

    launch_gemm(stream, 1, false, true, false, X, w_rep_l3t, in(21), nullptr, T1, nullptr, B, 1024, 1536);
    launch_ln(stream, T1, in(27), in(28), X, B, 1024);

    launch_ln(stream, X, in(55), in(56), T1, B, 1024);                                   // rb3
    launch_gemm(stream, 1, false, true, false, T1, w_rb3_fc1t, in(51), nullptr, T2, nullptr, B, 2048, 1024);
    launch_ln(stream, T2, in(57), in(58), T1, B, 2048);
    launch_gemm(stream, 0, true, true, false, T1, w_rb3_fc2t, in(53), X, X, nullptr, B, 1024, 2048);

    // ---- rep MoE (dense experts, top-2 combine + residual) ----
    gate_kernel<<<dim3(B / 8), dim3(256), 0, stream>>>((const __bf16*)X, in(32), in(31), GL, 1024);
    for (int e = 0; e < 4; ++e) {
        launch_gemm(stream, 1, false, true, false, X,
                    (char*)w_rmoe_w1t + (size_t)e * 1536 * 1024 * 2, in(29) + e * 1536, nullptr,
                    (char*)EH + (size_t)e * B * 1536 * 2, nullptr, B, 1536, 1024);
        launch_gemm(stream, 0, false, true, false, (char*)EH + (size_t)e * B * 1536 * 2,
                    (char*)w_rmoe_w2t + (size_t)e * 1024 * 1536 * 2, in(30) + e * 1024, nullptr,
                    (char*)EY + (size_t)e * B * 1024 * 2, nullptr, B, 1024, 1536);
    }
    moe_combine_kernel<<<dim3((B * 1024) / 256), dim3(256), 0, stream>>>(
        GL, (const __bf16*)EY, (const __bf16*)X, (__bf16*)X, B, 1024);

    // ---- rb4 -> h (bf16 in H, fp32 straight into d_out) ----
    float* out_f = (float*)d_out;
    float* h_f32 = out_f + (size_t)B * 64 + B;
    launch_ln(stream, X, in(63), in(64), T1, B, 1024);
    launch_gemm(stream, 1, false, true, false, T1, w_rb4_fc1t, in(59), nullptr, T2, nullptr, B, 2048, 1024);
    launch_ln(stream, T2, in(65), in(66), T1, B, 2048);
    launch_gemm(stream, 0, true, true, true, T1, w_rb4_fc2t, in(61), X, H, h_f32, B, 1024, 2048);

    // ---- policy MoE on h ----
    gate_kernel<<<dim3(B / 8), dim3(256), 0, stream>>>((const __bf16*)H, in(14), in(13), GL, 1024);
    for (int e = 0; e < 4; ++e) {
        launch_gemm(stream, 1, false, true, false, H,
                    (char*)w_pmoe_w1t + (size_t)e * 1024 * 1024 * 2, in(11) + e * 1024, nullptr,
                    (char*)EH + (size_t)e * B * 1024 * 2, nullptr, B, 1024, 1024);
        launch_gemm(stream, 0, false, true, false, (char*)EH + (size_t)e * B * 1024 * 2,
                    (char*)w_pmoe_w2t + (size_t)e * 1024 * 1024 * 2, in(12) + e * 1024, nullptr,
                    (char*)EY + (size_t)e * B * 1024 * 2, nullptr, B, 1024, 1024);
    }
    moe_combine_kernel<<<dim3((B * 1024) / 256), dim3(256), 0, stream>>>(
        GL, (const __bf16*)EY, (const __bf16*)H, (__bf16*)X, B, 1024);

    // ---- policy head ----
    launch_gemm(stream, 1, false, true, false, X, w_pol_l1t, in(1), nullptr, T1, nullptr, B, 512, 1024);
    launch_ln(stream, T1, in(7), in(8), T2, B, 512);
    launch_gemm(stream, 1, false, true, false, T2, w_pol_l2t, in(3), nullptr, T1, nullptr, B, 256, 512);
    launch_ln(stream, T1, in(9), in(10), T2, B, 256);
    launch_gemm(stream, 0, false, false, true, T2, w_pol_l3t, in(5), nullptr, nullptr, out_f, B, 64, 256);

    // ---- value head ----
    launch_gemm(stream, 1, false, true, false, H, w_val_l1t, in(67), nullptr, T1, nullptr, B, 512, 1024);
    launch_ln(stream, T1, in(73), in(74), T2, B, 512);
    launch_gemm(stream, 1, false, true, false, T2, w_val_l2t, in(69), nullptr, T1, nullptr, B, 256, 512);
    launch_ln(stream, T1, in(75), in(76), T2, B, 256);
    value_head_kernel<<<dim3(B / 8), dim3(256), 0, stream>>>(
        (const __bf16*)T2, in(72), in(71), out_f + (size_t)B * 64, 256);
}
